// SpectralPointMixing_17858474017207
// MI455X (gfx1250) — matrix operand order repaired and checked by reading, not compiled, not run on hardware
//
#include <hip/hip_runtime.h>
#include <hip/hip_bf16.h>

typedef __attribute__((ext_vector_type(16))) _Float16 v16h;
typedef __attribute__((ext_vector_type(2)))  _Float16 v2h;
typedef __attribute__((ext_vector_type(8)))  float    v8f;

#define BATCH 8
#define NPTS  65536
#define CDIM  64
#define MDIM  64

static __device__ __forceinline__ unsigned pack2h(float a, float b) {
    union { v2h h; unsigned u; } p;
    p.h.x = (_Float16)a;
    p.h.y = (_Float16)b;
    return p.u;
}

// ---- kernel 1: projection  coeff[b,m,c] = sum_n basis(n,m)*inputs[n,c] ----
// grid = BATCH * K1_NBLK blocks of 128 threads (4 waves); wave w owns m-tile w.
#define K1_NBLK  32                 // splits of N (reduced via f32 atomics)
#define K1_NPW   (NPTS / K1_NBLK)   // 2048 points per workgroup
#define K1_STEPS (K1_NPW / 32)      // 64 K-steps of 32

__global__ __launch_bounds__(128) void spectral_project(
    const float* __restrict__ inputs,   // [B,N,C]
    const float* __restrict__ coords,   // [B,N,2]
    const float* __restrict__ freqs,    // [M,2]
    float* __restrict__ coeffR,         // [B,M,C]  (sum cos * in)
    float* __restrict__ coeffI)         // [B,M,C]  (sum sin * in)
{
    __shared__ _Float16 lin[2][CDIM * 34];  // inputs chunk, transposed [c][k], pad->34
    __shared__ float    lco[2][64];         // coords chunk (x,y interleaved)

    const int tid  = threadIdx.x;
    const int wave = tid >> 5;            // m-tile 0..3
    const int lane = tid & 31;
    const int g    = lane >> 4;           // half-wave group
    const int lm   = lane & 15;

    const int b      = blockIdx.x / K1_NBLK;
    const int n0base = (blockIdx.x % K1_NBLK) * K1_NPW;

    const int   m  = wave * 16 + lm;      // this lane's A-matrix row (mode)
    const float kx = freqs[2 * m + 0];
    const float ky = freqs[2 * m + 1];

    // per-thread staging task: pairs of rows (k,k+1) x 4 consecutive c
    // task j in {tid, tid+128}: kp=j/16 -> k=2*kp ; cg=j%16 -> c=4*cg
    const int kp0 = tid >> 4, cg0 = tid & 15;          // task 0 (task 1: kp0+8)

    const v8f z = {0.f, 0.f, 0.f, 0.f, 0.f, 0.f, 0.f, 0.f};
    v8f accR[4] = {z, z, z, z};
    v8f accI[4] = {z, z, z, z};

    float4 ra0, ra1, rb0, rb1;   // prefetch registers (rows k / k+1, two tasks)
    float2 rco;                  // coords prefetch (tid < 32)

    auto issue_loads = [&](int s) {
        const float* src = inputs + ((size_t)b * NPTS + n0base + s * 32) * CDIM;
        const float4* s4 = (const float4*)src;
        ra0 = s4[(2 * kp0) * 16 + cg0];       // task 0, row k
        ra1 = s4[(2 * kp0 + 1) * 16 + cg0];   // task 0, row k+1
        rb0 = s4[(2 * kp0 + 16) * 16 + cg0];  // task 1, row k
        rb1 = s4[(2 * kp0 + 17) * 16 + cg0];  // task 1, row k+1
        if (tid < 32)
            rco = ((const float2*)(coords + ((size_t)b * NPTS + n0base + s * 32) * 2))[tid];
    };

    auto store_lds = [&](int buf) {
        // pack (k,k+1) halfword pairs -> aligned b32 LDS stores
        unsigned* d32 = (unsigned*)&lin[buf][0];
        const int base0 = (4 * cg0) * 17 + kp0;        // (c*34 + k)/2 with k=2*kp
        const int base1 = base0 + 8;
        d32[base0 + 0 * 17] = pack2h(ra0.x, ra1.x);
        d32[base0 + 1 * 17] = pack2h(ra0.y, ra1.y);
        d32[base0 + 2 * 17] = pack2h(ra0.z, ra1.z);
        d32[base0 + 3 * 17] = pack2h(ra0.w, ra1.w);
        d32[base1 + 0 * 17] = pack2h(rb0.x, rb1.x);
        d32[base1 + 1 * 17] = pack2h(rb0.y, rb1.y);
        d32[base1 + 2 * 17] = pack2h(rb0.z, rb1.z);
        d32[base1 + 3 * 17] = pack2h(rb0.w, rb1.w);
        if (tid < 32) ((float2*)&lco[buf][0])[tid] = rco;
    };

    issue_loads(0);
    int buf = 0;
    for (int s = 0; s < K1_STEPS; ++s) {
        store_lds(buf);                             // consumes in-flight loads
        if (s + 1 < K1_STEPS) issue_loads(s + 1);   // overlap next chunk with compute
        __syncthreads();

        // A fragments: basis values in-register (v_cos/v_sin take revolutions)
        v16h ac, as;
        #pragma unroll
        for (int e = 0; e < 16; ++e) {
            const int k = 16 * g + e;
            const float x = lco[buf][2 * k + 0], y = lco[buf][2 * k + 1];
            const float t = kx * x + ky * y;          // phase / 2pi
            ac[e] = (_Float16)__builtin_amdgcn_cosf(t);
            as[e] = (_Float16)__builtin_amdgcn_sinf(t);
        }

        #pragma unroll
        for (int ct = 0; ct < 4; ++ct) {
            // B fragment: 16 contiguous halfwords per lane from transposed LDS
            union { v16h h; unsigned u[8]; } bf;
            const unsigned* p =
                (const unsigned*)&lin[buf][(ct * 16 + lm) * 34 + 16 * g];
            #pragma unroll
            for (int j = 0; j < 8; ++j) bf.u[j] = p[j];
            accR[ct] = __builtin_amdgcn_wmma_f32_16x16x32_f16(
                false, ac, false, bf.h, (short)0, accR[ct], false, false);
            accI[ct] = __builtin_amdgcn_wmma_f32_16x16x32_f16(
                false, as, false, bf.h, (short)0, accI[ct], false, false);
        }
        buf ^= 1;
    }

    // reduce N-splits: native f32 global atomics (64K distinct addrs, L2-resident)
    #pragma unroll
    for (int ct = 0; ct < 4; ++ct) {
        #pragma unroll
        for (int r = 0; r < 8; ++r) {
            const int mm = wave * 16 + r + 8 * g;
            const int cc = ct * 16 + lm;
            const size_t o = ((size_t)b * MDIM + mm) * CDIM + cc;
            unsafeAtomicAdd(&coeffR[o], accR[ct][r]);
            unsafeAtomicAdd(&coeffI[o], accI[ct][r]);
        }
    }
}

// ---- kernel 2: apply complex weight, 1/N, emit f16 mixed matrices ----
__global__ __launch_bounds__(256) void spectral_mix(
    const float* __restrict__ coeffR, const float* __restrict__ coeffI,
    const float* __restrict__ wR, const float* __restrict__ wI,
    _Float16* __restrict__ mixR, _Float16* __restrict__ mixI)
{
    const int idx = blockIdx.x * 256 + threadIdx.x;   // B*M*C = 32768
    const int mc  = idx & (MDIM * CDIM - 1);
    const float cr = coeffR[idx];
    const float S  = coeffI[idx];                     // stored +sum(sin*in); ci = -S/N
    const float wr = wR[mc], wi = wI[mc];
    const float invN = 1.0f / (float)NPTS;
    mixR[idx] = (_Float16)((cr * wr + S * wi) * invN);
    mixI[idx] = (_Float16)((cr * wi - S * wr) * invN);
}

// ---- kernel 3: reconstruction  out[n,c] = sum_m cos*mixR - sin*mixI ----
#define K2_TPB 64    // 16-row n-tiles per workgroup (4 waves x 16 iters)

__global__ __launch_bounds__(128) void spectral_recon(
    const float* __restrict__ coords, const float* __restrict__ freqs,
    const _Float16* __restrict__ mixR, const _Float16* __restrict__ mixI,
    float* __restrict__ out)
{
    __shared__ _Float16 lmr[CDIM * 66];   // mixed, transposed [c][m], pad->66
    __shared__ _Float16 lmi[CDIM * 66];
    __shared__ float    lfr[MDIM * 2];

    const int tid  = threadIdx.x;
    const int wave = tid >> 5;
    const int lane = tid & 31;
    const int g    = lane >> 4;
    const int lm   = lane & 15;

    const int blocksPerB = (NPTS / 16) / K2_TPB;      // 64
    const int b  = blockIdx.x / blocksPerB;
    const int t0 = (blockIdx.x % blocksPerB) * K2_TPB;

    const _Float16* mr = mixR + (size_t)b * MDIM * CDIM;
    const _Float16* mi = mixI + (size_t)b * MDIM * CDIM;
    for (int i = tid; i < MDIM * CDIM; i += 128) {
        const int m = i >> 6, c = i & 63;
        lmr[c * 66 + m] = mr[i];
        lmi[c * 66 + m] = mi[i];
    }
    if (tid < 128) lfr[tid] = freqs[tid];
    __syncthreads();

    // B fragments are loop-invariant: hoist all 16 (4 ctiles x 2 ksteps x re/im)
    v16h fbr[8], fbi[8];
    #pragma unroll
    for (int ks = 0; ks < 2; ++ks) {
        #pragma unroll
        for (int ct = 0; ct < 4; ++ct) {
            const int base = (ct * 16 + lm) * 66 + ks * 32 + 16 * g;
            union { v16h h; unsigned u[8]; } br, bi;
            const unsigned* pr = (const unsigned*)&lmr[base];
            const unsigned* pi = (const unsigned*)&lmi[base];
            #pragma unroll
            for (int j = 0; j < 8; ++j) { br.u[j] = pr[j]; bi.u[j] = pi[j]; }
            fbr[ks * 4 + ct] = br.h;
            fbi[ks * 4 + ct] = bi.h;
        }
    }

    // coords prefetch for the first tile
    float2 pc = *(const float2*)&coords[((size_t)b * NPTS +
                                         (size_t)(t0 + wave) * 16 + lm) * 2];

    for (int it = 0; it < K2_TPB / 4; ++it) {
        const int nt = t0 + it * 4 + wave;            // 16-row tile index
        const float x = pc.x, y = pc.y;
        if (it + 1 < K2_TPB / 4)
            pc = *(const float2*)&coords[((size_t)b * NPTS +
                    (size_t)(t0 + (it + 1) * 4 + wave) * 16 + lm) * 2];

        const v8f z = {0.f, 0.f, 0.f, 0.f, 0.f, 0.f, 0.f, 0.f};
        v8f acc[4] = {z, z, z, z};

        #pragma unroll
        for (int ks = 0; ks < 2; ++ks) {              // K = m, two steps of 32
            v16h ac, ns;
            #pragma unroll
            for (int e = 0; e < 16; ++e) {
                const int mk = ks * 32 + 16 * g + e;
                const float t = lfr[2 * mk] * x + lfr[2 * mk + 1] * y;
                ac[e] = (_Float16)__builtin_amdgcn_cosf(t);
                ns[e] = (_Float16)(-__builtin_amdgcn_sinf(t)); // f16 WMMA: no A-neg
            }
            #pragma unroll
            for (int ct = 0; ct < 4; ++ct) {
                acc[ct] = __builtin_amdgcn_wmma_f32_16x16x32_f16(
                    false, ac, false, fbr[ks * 4 + ct], (short)0, acc[ct], false, false);
                acc[ct] = __builtin_amdgcn_wmma_f32_16x16x32_f16(
                    false, ns, false, fbi[ks * 4 + ct], (short)0, acc[ct], false, false);
            }
        }

        float* dst = out + ((size_t)b * NPTS + (size_t)nt * 16) * CDIM;
        #pragma unroll
        for (int ct = 0; ct < 4; ++ct)
            #pragma unroll
            for (int r = 0; r < 8; ++r)
                dst[(r + 8 * g) * CDIM + ct * 16 + lm] = acc[ct][r];
    }
}

extern "C" void kernel_launch(void* const* d_in, const int* in_sizes, int n_in,
                              void* d_out, int out_size, void* d_ws, size_t ws_size,
                              hipStream_t stream) {
    const float* inputs = (const float*)d_in[0];   // [B,N,C]
    const float* coords = (const float*)d_in[1];   // [B,N,2]
    const float* wR     = (const float*)d_in[2];   // [M,C]
    const float* wI     = (const float*)d_in[3];   // [M,C]
    const float* freqs  = (const float*)d_in[4];   // [M,2]
    float* out = (float*)d_out;

    const size_t nCoef = (size_t)BATCH * MDIM * CDIM;      // 32768
    float*    coeffR = (float*)d_ws;
    float*    coeffI = coeffR + nCoef;
    _Float16* mixR   = (_Float16*)(coeffI + nCoef);
    _Float16* mixI   = mixR + nCoef;

    // zero the atomic accumulators each launch (capture-legal)
    (void)hipMemsetAsync(d_ws, 0, 2 * nCoef * sizeof(float), stream);

    spectral_project<<<BATCH * K1_NBLK, 128, 0, stream>>>(
        inputs, coords, freqs, coeffR, coeffI);
    spectral_mix<<<(int)(nCoef / 256), 256, 0, stream>>>(
        coeffR, coeffI, wR, wI, mixR, mixI);
    spectral_recon<<<BATCH * ((NPTS / 16) / K2_TPB), 128, 0, stream>>>(
        coords, freqs, mixR, mixI, out);
}
